// DenseGIN_21045339750900
// MI455X (gfx1250) — matrix operand order repaired and checked by reading, not compiled, not run on hardware
//
#include <hip/hip_runtime.h>
#include <stdint.h>

// ---------------------------------------------------------------------------
// Types
// ---------------------------------------------------------------------------
typedef __bf16 bf16_t;
typedef __attribute__((ext_vector_type(16))) __bf16 v16bf;
typedef __attribute__((ext_vector_type(8)))  __bf16 v8bf;
typedef __attribute__((ext_vector_type(8)))  float  v8f;

#define GIN_IN_C  128
#define GIN_HID   256
#define GIN_OUT_C 40
#define GIN_BN_EPS 1e-5f

__device__ __forceinline__ bf16_t f2bf(float f) {
  uint32_t u = __builtin_bit_cast(uint32_t, f);
  u += 0x7FFFu + ((u >> 16) & 1u);           // round-to-nearest-even
  unsigned short h = (unsigned short)(u >> 16);
  return __builtin_bit_cast(bf16_t, h);
}

// ---------------------------------------------------------------------------
// Elementwise helpers
// ---------------------------------------------------------------------------
__global__ void fill_zero_f32(float* __restrict__ p, int n) {
  int i = blockIdx.x * blockDim.x + threadIdx.x;
  if (i < n) p[i] = 0.0f;
}

// wt[n*din_p + k] = bf16(w[k*dout + n]), zero padded  (B^T layout for WMMA)
__global__ void prep_weight_t(const float* __restrict__ w, bf16_t* __restrict__ wt,
                              int din, int dout, int din_p, int dout_p) {
  int i = blockIdx.x * blockDim.x + threadIdx.x;
  int tot = din_p * dout_p;
  if (i >= tot) return;
  int n = i / din_p;
  int k = i - n * din_p;
  float v = (n < dout && k < din) ? w[(size_t)k * dout + n] : 0.0f;
  wt[i] = f2bf(v);
}

// agg[dst] += ew * x[src]   (edge-parallel, float4 gather, f32 L2 atomics)
__global__ void scatter_add_f32(const float* __restrict__ x,
                                const int* __restrict__ src,
                                const int* __restrict__ dst,
                                const float* __restrict__ ew,
                                float* __restrict__ agg, int C, int E) {
  int i = blockIdx.x * blockDim.x + threadIdx.x;
  int c4n = C >> 2;
  int total = E * c4n;
  if (i >= total) return;
  int e = i / c4n;
  int c = (i - e * c4n) << 2;
  int s = src[e], d = dst[e];
  float w = ew[e];
  const float4 xv = *(const float4*)(x + (size_t)s * C + c);
  float* o = agg + (size_t)d * C + c;
  unsafeAtomicAdd(o + 0, w * xv.x);
  unsafeAtomicAdd(o + 1, w * xv.y);
  unsafeAtomicAdd(o + 2, w * xv.z);
  unsafeAtomicAdd(o + 3, w * xv.w);
}

// o = bf16(a + b)
__global__ void add_cvt_bf16(const float* __restrict__ a, const float* __restrict__ b,
                             bf16_t* __restrict__ o, int n) {
  int i = blockIdx.x * blockDim.x + threadIdx.x;
  if (i < n) o[i] = f2bf(a[i] + b[i]);
}

// ---------------------------------------------------------------------------
// WMMA GEMM: D[M x Np] = A[M x Kp] * B[Kp x Np] (+bias, fused epilogue)
//   A:  bf16 row-major (stride Kp)
//   Bt: bf16 B-transposed (Np x Kp row-major), Np multiple of 64
//   One wave computes a 16x64 output slab (1x4 register tiling). Per K-step
//   ALL loads (1 A fragment + 4 B fragments) are issued before the 4 WMMAs so
//   the scheduler can use partial s_wait_loadcnt and overlap loads with WMMA.
//   mode 0: ReLU(acc+bias)           -> bf16 out (stride ldo)
//   mode 1: BN(acc+bias) then ReLU   -> f32 out  (stride ldo)
//   mode 2: acc+bias (cols < nvalid) -> f32 out  (stride ldo)
// ---------------------------------------------------------------------------
__global__ __launch_bounds__(256)
void gin_gemm_wmma(const bf16_t* __restrict__ A,
                   const bf16_t* __restrict__ Bt,
                   const float* __restrict__ bias,
                   int M, int Kp, int Np, int nvalid, int mode,
                   bf16_t* __restrict__ outB, float* __restrict__ outF, int ldo,
                   const float* __restrict__ bn_g, const float* __restrict__ bn_b,
                   const float* __restrict__ bn_m, const float* __restrict__ bn_v) {
  const int lane = threadIdx.x & 31;
  const int wid  = (blockIdx.x << 3) + (threadIdx.x >> 5);
  const int groupsN = Np >> 6;         // groups of 4 N-tiles (64 cols)
  const int tilesM  = M >> 4;          // M multiple of 16 (50000 = 3125*16)
  const int tm  = wid / groupsN;
  const int tng = wid - tm * groupsN;
  if (tm >= tilesM) return;

  const int hlf = lane >> 4;           // 0: lanes 0-15, 1: lanes 16-31
  const int l15 = lane & 15;

  // A 16-bit 16x32 layout: lo lanes K {0..7,16..23}, hi lanes K {8..15,24..31}
  const bf16_t* ap = A + (size_t)(tm * 16 + l15) * Kp + hlf * 8;
  // B 16-bit 32x16 layout: lo lanes K {0..7,16..23}, hi lanes K {8..15,24..31}
  const bf16_t* bp = Bt + (size_t)(tng * 64 + l15) * Kp + hlf * 8;

  v8f acc[4] = {};
#pragma unroll 2
  for (int k = 0; k < Kp; k += 32) {
    // --- issue every load for this K-step first ---
    v8bf a0 = *(const v8bf*)(ap + k);
    v8bf a1 = *(const v8bf*)(ap + k + 16);
    v8bf b0[4], b1[4];
#pragma unroll
    for (int t = 0; t < 4; ++t) {
      const bf16_t* bc = bp + (size_t)t * 16 * Kp + k;
      b0[t] = *(const v8bf*)(bc);
      b1[t] = *(const v8bf*)(bc + 16);
    }
    // --- then 4 back-to-back WMMAs sharing the A fragment ---
    v16bf av;
#pragma unroll
    for (int j = 0; j < 8; ++j) { av[j] = a0[j]; av[j + 8] = a1[j]; }
#pragma unroll
    for (int t = 0; t < 4; ++t) {
      v16bf bv;
#pragma unroll
      for (int j = 0; j < 8; ++j) { bv[j] = b0[t][j]; bv[j + 8] = b1[t][j]; }
      acc[t] = __builtin_amdgcn_wmma_f32_16x16x32_bf16(false, av, false, bv,
                                                       (short)0, acc[t], false, false);
    }
  }

  // C/D layout: VGPR j -> row (j + 8*hlf), col = l15 within each 16-wide tile
  const int rowbase = tm * 16 + hlf * 8;
  const int colbase = tng * 64 + l15;

#pragma unroll
  for (int t = 0; t < 4; ++t) {
    const int col = colbase + t * 16;
    const float bias_v = (col < nvalid) ? bias[col] : 0.0f;
    if (mode == 0) {                                  // ReLU -> bf16
#pragma unroll
      for (int j = 0; j < 8; ++j) {
        float v = fmaxf(acc[t][j] + bias_v, 0.0f);
        outB[(size_t)(rowbase + j) * ldo + col] = f2bf(v);
      }
    } else if (mode == 1) {                           // BN + ReLU -> f32
      const float scale = bn_g[col] * rsqrtf(bn_v[col] + GIN_BN_EPS);
      const float mu = bn_m[col], beta = bn_b[col];
#pragma unroll
      for (int j = 0; j < 8; ++j) {
        float v = (acc[t][j] + bias_v - mu) * scale + beta;
        outF[(size_t)(rowbase + j) * ldo + col] = fmaxf(v, 0.0f);
      }
    } else {                                          // plain -> f32 (guarded)
      if (col < nvalid) {
#pragma unroll
        for (int j = 0; j < 8; ++j)
          outF[(size_t)(rowbase + j) * ldo + col] = acc[t][j] + bias_v;
      }
    }
  }
}

// ---------------------------------------------------------------------------
// Row-wise log-softmax (C = 40)
// ---------------------------------------------------------------------------
__global__ void log_softmax_rows(const float* __restrict__ in, float* __restrict__ out,
                                 int M, int C) {
  int r = blockIdx.x * blockDim.x + threadIdx.x;
  if (r >= M) return;
  const float* p = in + (size_t)r * C;
  float mx = p[0];
  for (int c = 1; c < C; ++c) mx = fmaxf(mx, p[c]);
  float s = 0.0f;
  for (int c = 0; c < C; ++c) s += __expf(p[c] - mx);
  float l = mx + __logf(s);
  float* q = out + (size_t)r * C;
  for (int c = 0; c < C; ++c) q[c] = p[c] - l;
}

// ---------------------------------------------------------------------------
// Host driver
// ---------------------------------------------------------------------------
extern "C" void kernel_launch(void* const* d_in, const int* in_sizes, int n_in,
                              void* d_out, int out_size, void* d_ws, size_t ws_size,
                              hipStream_t stream) {
  const int N = in_sizes[0] / GIN_IN_C;   // 50000
  const int E = in_sizes[2];              // 800000

  const float* x  = (const float*)d_in[0];
  const int*   ei = (const int*)d_in[1];
  const float* ew = (const float*)d_in[2];
  const int* src = ei;
  const int* dst = ei + E;

  const float* w1[3] = {(const float*)d_in[3], (const float*)d_in[7],  (const float*)d_in[11]};
  const float* b1[3] = {(const float*)d_in[4], (const float*)d_in[8],  (const float*)d_in[12]};
  const float* w2[3] = {(const float*)d_in[5], (const float*)d_in[9],  (const float*)d_in[13]};
  const float* b2[3] = {(const float*)d_in[6], (const float*)d_in[10], (const float*)d_in[14]};
  const float* bng[2] = {(const float*)d_in[15], (const float*)d_in[19]};
  const float* bnb[2] = {(const float*)d_in[16], (const float*)d_in[20]};
  const float* bnm[2] = {(const float*)d_in[17], (const float*)d_in[21]};
  const float* bnv[2] = {(const float*)d_in[18], (const float*)d_in[22]};

  // Workspace carve (256-B aligned). Peak ~155 MB — entirely L2-resident.
  uint8_t* wp = (uint8_t*)d_ws;
  auto carve = [&](size_t bytes) -> void* {
    void* r = (void*)wp;
    wp += (bytes + 255) & ~(size_t)255;
    return r;
  };
  float*  agg    = (float*) carve((size_t)N * GIN_HID * 4);   // aggregation
  bf16_t* hin    = (bf16_t*)carve((size_t)N * GIN_HID * 2);   // bf16(agg + x)
  bf16_t* hmid   = (bf16_t*)carve((size_t)N * GIN_HID * 2);   // MLP hidden (bf16)
  float*  xcur   = (float*) carve((size_t)N * GIN_HID * 4);   // layer output f32
  float*  logits = (float*) carve((size_t)N * GIN_OUT_C * 4);
  bf16_t* w1t0 = (bf16_t*)carve((size_t)256 * 128 * 2);
  bf16_t* w2t0 = (bf16_t*)carve((size_t)256 * 256 * 2);
  bf16_t* w1t1 = (bf16_t*)carve((size_t)256 * 256 * 2);
  bf16_t* w2t1 = (bf16_t*)carve((size_t)256 * 256 * 2);
  bf16_t* w1t2 = (bf16_t*)carve((size_t)64  * 256 * 2);       // dout 40 -> pad 64
  bf16_t* w2t2 = (bf16_t*)carve((size_t)64  * 64  * 2);       // 40x40  -> pad 64x64

  auto prep = [&](const float* w, bf16_t* wt, int din, int dout, int din_p, int dout_p) {
    int tot = din_p * dout_p;
    prep_weight_t<<<(tot + 255) / 256, 256, 0, stream>>>(w, wt, din, dout, din_p, dout_p);
  };
  prep(w1[0], w1t0, 128, 256, 128, 256);
  prep(w2[0], w2t0, 256, 256, 256, 256);
  prep(w1[1], w1t1, 256, 256, 256, 256);
  prep(w2[1], w2t1, 256, 256, 256, 256);
  prep(w1[2], w1t2, 256, 40, 256, 64);
  prep(w2[2], w2t2, 40,  40, 64,  64);

  auto gemm = [&](const bf16_t* A, const bf16_t* Bt, const float* bias,
                  int Kp, int Np, int nvalid, int mode,
                  bf16_t* outB, float* outF, int ldo,
                  const float* g, const float* b, const float* m, const float* v) {
    int tiles = (N / 16) * (Np / 64);   // one wave per 16x64 slab
    gin_gemm_wmma<<<(tiles + 7) / 8, 256, 0, stream>>>(
        A, Bt, bias, N, Kp, Np, nvalid, mode, outB, outF, ldo, g, b, m, v);
  };

  const float* xsrc = x;
  for (int li = 0; li < 3; ++li) {
    const int C  = (li == 0) ? GIN_IN_C : GIN_HID;
    const int nc = N * C;
    fill_zero_f32<<<(nc + 255) / 256, 256, 0, stream>>>(agg, nc);
    int st = E * (C / 4);
    scatter_add_f32<<<(st + 255) / 256, 256, 0, stream>>>(xsrc, src, dst, ew, agg, C, E);
    add_cvt_bf16<<<(nc + 255) / 256, 256, 0, stream>>>(agg, xsrc, hin, nc);

    if (li == 0) {
      gemm(hin,  w1t0, b1[0], 128, 256, 256, 0, hmid, nullptr, 256,
           nullptr, nullptr, nullptr, nullptr);
      gemm(hmid, w2t0, b2[0], 256, 256, 256, 1, nullptr, xcur, 256,
           bng[0], bnb[0], bnm[0], bnv[0]);
      xsrc = xcur;
    } else if (li == 1) {
      gemm(hin,  w1t1, b1[1], 256, 256, 256, 0, hmid, nullptr, 256,
           nullptr, nullptr, nullptr, nullptr);
      gemm(hmid, w2t1, b2[1], 256, 256, 256, 1, nullptr, xcur, 256,
           bng[1], bnb[1], bnm[1], bnv[1]);
      xsrc = xcur;
    } else {
      // dout 40 padded to 64; padded lanes are exactly zero through the MLP
      // (zero weights + zero bias + ReLU), mode-2 write is guarded to col<40.
      gemm(hin,  w1t2, b1[2], 256, 64, 40, 0, hmid, nullptr, 64,
           nullptr, nullptr, nullptr, nullptr);
      gemm(hmid, w2t2, b2[2], 64,  64, 40, 2, nullptr, logits, 40,
           nullptr, nullptr, nullptr, nullptr);
    }
  }

  log_softmax_rows<<<(N + 255) / 256, 256, 0, stream>>>(logits, (float*)d_out, N, GIN_OUT_C);
  (void)n_in; (void)out_size; (void)ws_size;
}
